// EncoderGRU_25048249270324
// MI455X (gfx1250) — matrix operand order repaired and checked by reading, not compiled, not run on hardware
//
#include <hip/hip_runtime.h>
#include <hip/hip_bf16.h>
#include <math.h>

#define SEQ   4096
#define HID   1024
#define EMB   1024
#define G3    3072          // 3*HID
#define AROWS 4098          // 4096 tokens + SOS row + EOS row

typedef __attribute__((ext_vector_type(16))) __bf16 v16bf;
typedef __attribute__((ext_vector_type(8)))  float  v8f;
typedef __attribute__((ext_vector_type(4)))  unsigned int u32x4;
typedef __attribute__((ext_vector_type(8)))  int i32x8;
typedef __attribute__((ext_vector_type(4)))  int i32x4;

union Frag16 { v16bf v; unsigned int u[8]; };

// Pack two f32 -> packed bf16 dword (compiler lowers to v_cvt_pk_bf16_f32)
__device__ inline unsigned int pk2(float lo, float hi) {
  unsigned short l = __builtin_bit_cast(unsigned short, (__bf16)lo);
  unsigned short h = __builtin_bit_cast(unsigned short, (__bf16)hi);
  return (unsigned int)l | ((unsigned int)h << 16);
}

// ---------------------------------------------------------------------------
// Kernel 0: zero the grid-barrier counters (workspace is poisoned by harness)
// ---------------------------------------------------------------------------
__global__ void init_kernel(int* cnt) {
  cnt[threadIdx.x] = 0;
}

// ---------------------------------------------------------------------------
// Kernel 1: Gi[gm,:] = bf16(X[gm,:]) @ bf16(w_ih)^T + b_ih     (f32 accum)
//   Block 256 thr (8 waves), tile 64(M) x 128(N), K-step 32, software-pipelined
//   branch-free float4 staging. Wave (wm 0..3, wn 0..1): 1 A-frag + 4 B-frags,
//   4 x v_wmma_f32_16x16x32_bf16 per K-step.
// ---------------------------------------------------------------------------
#define BM 64
#define BN 128
#define BK 32
#define LDA (BK + 2)   // 34 bf16 = 17 dwords/row (odd -> conflict-free b32)
#define LDB (BK + 2)

__global__ __launch_bounds__(256) void gi_gemm_kernel(
    const int* __restrict__ tokens, const float* __restrict__ emb,
    const float* __restrict__ w_ih, const float* __restrict__ b_ih,
    float* __restrict__ Gi)
{
  __shared__ __bf16 As[BM * LDA];   // As[m][k]
  __shared__ __bf16 Bs[BN * LDB];   // Bs[n][k] = w_ih[gn][k]

  const int tid  = threadIdx.x;
  const int lane = tid & 31;
  const int wid  = tid >> 5;
  const int wm   = wid >> 1;        // 0..3
  const int wn   = wid & 1;         // 0..1
  const int bM   = blockIdx.y;
  const int bN   = blockIdx.x;

  // A staging: row = tid/4 (0..63), 8 consecutive k at (tid%4)*8
  const int ldRowA = tid >> 2;
  const int kcA    = (tid & 3) * 8;
  // B staging: row = tid/2 (0..127), 16 consecutive k at (tid%2)*16
  const int ldRowB = tid >> 1;
  const int kcB    = (tid & 1) * 16;

  // Branch-free gather: invalid rows read emb[0] scaled by 0.
  const int gmRow = bM * BM + ldRowA;
  int tok = 0;
  float scaleA = 1.0f;
  if (gmRow < SEQ)            tok = tokens[gmRow];
  else if (gmRow == SEQ)      tok = 0;          // SOS
  else if (gmRow == SEQ + 1)  tok = 1;          // EOS
  else                        scaleA = 0.0f;
  const float* aSrc = emb + (size_t)tok * EMB + kcA;
  const float* bSrc = w_ih + (size_t)(bN * BN + ldRowB) * EMB + kcB;

  v8f acc[4] = {};

  // Prologue loads (k0 = 0)
  float4 sa0 = *(const float4*)(aSrc + 0);
  float4 sa1 = *(const float4*)(aSrc + 4);
  float4 sb0 = *(const float4*)(bSrc + 0);
  float4 sb1 = *(const float4*)(bSrc + 4);
  float4 sb2 = *(const float4*)(bSrc + 8);
  float4 sb3 = *(const float4*)(bSrc + 12);

  for (int k0 = 0; k0 < EMB; k0 += BK) {
    __syncthreads();
    // ---- store staged tile to LDS as packed bf16 (ds_store_b32) ----
    unsigned int* arow = (unsigned int*)(As + ldRowA * LDA + kcA);
    arow[0] = pk2(sa0.x * scaleA, sa0.y * scaleA);
    arow[1] = pk2(sa0.z * scaleA, sa0.w * scaleA);
    arow[2] = pk2(sa1.x * scaleA, sa1.y * scaleA);
    arow[3] = pk2(sa1.z * scaleA, sa1.w * scaleA);
    unsigned int* brow = (unsigned int*)(Bs + ldRowB * LDB + kcB);
    brow[0] = pk2(sb0.x, sb0.y);  brow[1] = pk2(sb0.z, sb0.w);
    brow[2] = pk2(sb1.x, sb1.y);  brow[3] = pk2(sb1.z, sb1.w);
    brow[4] = pk2(sb2.x, sb2.y);  brow[5] = pk2(sb2.z, sb2.w);
    brow[6] = pk2(sb3.x, sb3.y);  brow[7] = pk2(sb3.z, sb3.w);

    // ---- issue next K-tile's global loads (overlap with WMMA below) ----
    const bool more = (k0 + BK) < EMB;
    if (more) {
      const float* an = aSrc + k0 + BK;
      const float* bn = bSrc + k0 + BK;
      sa0 = *(const float4*)(an + 0);
      sa1 = *(const float4*)(an + 4);
      sb0 = *(const float4*)(bn + 0);
      sb1 = *(const float4*)(bn + 4);
      sb2 = *(const float4*)(bn + 8);
      sb3 = *(const float4*)(bn + 12);
    }
    __syncthreads();

    // ---- fragments, contiguous k = 16*kh + s in A and B, + 4 WMMAs ----
    const int kh = lane >> 4;
    Frag16 a;
    const __bf16* ar = As + (wm * 16 + (lane & 15)) * LDA;
    #pragma unroll
    for (int j = 0; j < 4; ++j) {
      a.u[j]     = *(const unsigned int*)(ar + kh * 16 + 2 * j);
      a.u[4 + j] = *(const unsigned int*)(ar + kh * 16 + 8 + 2 * j);
    }
    #pragma unroll
    for (int nt = 0; nt < 4; ++nt) {
      Frag16 b;
      const __bf16* br = Bs + (wn * 64 + nt * 16 + (lane & 15)) * LDB;
      #pragma unroll
      for (int j = 0; j < 8; ++j)
        b.u[j] = *(const unsigned int*)(br + kh * 16 + 2 * j);
      acc[nt] = __builtin_amdgcn_wmma_f32_16x16x32_bf16(
          false, a.v, false, b.v, (short)0, acc[nt], false, false);
    }
  }

  // ---- store D (+ bias). VGPR r: M = r + 8*(lane>=16); N = lane&15 ----
  const int n16 = lane & 15;
  const int mh  = lane >> 4;
  const int gmBase = bM * BM + wm * 16 + mh * 8;
  #pragma unroll
  for (int nt = 0; nt < 4; ++nt) {
    const int gn = bN * BN + wn * 64 + nt * 16 + n16;
    const float bi = b_ih[gn];
    #pragma unroll
    for (int r = 0; r < 8; ++r) {
      int gm = gmBase + r;
      if (gm < AROWS) Gi[(size_t)gm * G3 + gn] = acc[nt][r] + bi;
    }
  }
}

// ---------------------------------------------------------------------------
// Kernel 2: persistent bidirectional GRU scan.
//   64 blocks: dir = blk/32, j = blk%32 owns h[j*32 .. j*32+31].
//   96 threads: t -> gate g=t/32, elem i=t%32, w_hh row g*1024+j*32+i.
//   w_hh slice (96x1024) bf16-resident in LDS (~193 KB of the 320 KB WGP LDS),
//   preloaded via TDM tensor_load_to_lds when available. h exchanged through
//   L2 each step; per-direction monotonic atomic barrier.
// ---------------------------------------------------------------------------
#define RWGS        32
#define RTHREADS    96
#define WROW_STRIDE 1028   // 514 dwords: b64 accesses cover all 64 banks once
#define WL_BYTES    (RTHREADS * WROW_STRIDE * 2)         // 197376
#define HS_OFF      WL_BYTES
#define GHS_OFF     (HS_OFF + HID * 4)                   // 201472
#define STAGE_OFF   (GHS_OFF + 128 * 4)                  // 201984 (16B aligned)
#define STAGE_ROWS  8
#define SCAN_LDS    (STAGE_OFF + STAGE_ROWS * HID * 4)   // + 32 KB staging

__global__ __launch_bounds__(RTHREADS) void gru_scan_kernel(
    const float* __restrict__ w_hh, const float* __restrict__ b_hh,
    const float* __restrict__ Gi, float* hbuf, int* cnt,
    float* __restrict__ dout)
{
  extern __shared__ char smem[];
  __bf16* wl  = (__bf16*)smem;
  float*  hs  = (float*)(smem + HS_OFF);
  float*  ghs = (float*)(smem + GHS_OFF);

  const int dir  = blockIdx.x >> 5;
  const int j    = blockIdx.x & 31;
  const int tid  = threadIdx.x;
  const int g    = tid / 32;
  const int i    = tid & 31;
  const int hidx = j * 32 + i;
  const int wrow = g * HID + hidx;
  const float bhh = b_hh[wrow];

  int* myCnt = cnt + dir * 32;
  float* hb0 = hbuf + (dir * 2 + 0) * HID;
  float* hb1 = hbuf + (dir * 2 + 1) * HID;

  // ---- Preload bf16 w_hh slice into LDS ----
#if defined(__gfx1250__) && __has_builtin(__builtin_amdgcn_tensor_load_to_lds)
  {
    float* stage = (float*)(smem + STAGE_OFF);
    for (int gate = 0; gate < 3; ++gate) {
      for (int c = 0; c < 4; ++c) {
        if (tid < 32) {   // one wave issues the DMA (EXEC ignored by TDM)
          const int grow0 = gate * HID + j * 32 + c * STAGE_ROWS;
          unsigned long long ga =
              (unsigned long long)(const void*)(w_hh + (size_t)grow0 * HID);
          u32x4 g0;
          g0.x = 1u;                                   // count=1, user mode
          g0.y = (unsigned int)STAGE_OFF;              // lds_addr (bytes)
          g0.z = (unsigned int)(ga & 0xffffffffu);     // global_addr lo
          g0.w = (unsigned int)((ga >> 32) & 0x1ffffffu) | (2u << 30); // type=2
          i32x8 g1;
          g1[0] = (int)(2u << 16);            // data_size=4B; mask=0; no flags
          g1[1] = (int)(1024u << 16);         // tensor_dim0 = 1024 (lo16)
          g1[2] = (int)(8u << 16);            // dim0 hi=0 | tensor_dim1=8 lo16
          g1[3] = (int)(1024u << 16);         // dim1 hi=0 | tile_dim0=1024
          g1[4] = (int)STAGE_ROWS;            // tile_dim1=8, tile_dim2=0
          g1[5] = 1024;                       // tensor_dim0_stride lo32
          g1[6] = 0;                          // stride0 hi | stride1 lo
          g1[7] = 0;
          i32x4 gz = {0, 0, 0, 0};
#if __clang_major__ >= 23
          i32x8 gz8 = {0, 0, 0, 0, 0, 0, 0, 0};
          __builtin_amdgcn_tensor_load_to_lds(g0, g1, gz, gz, gz8, 0);
#else
          __builtin_amdgcn_tensor_load_to_lds(g0, g1, gz, gz, 0);
#endif
          __builtin_amdgcn_s_wait_tensorcnt(0);
        }
        __syncthreads();
        // convert staged f32 rows -> resident bf16 rows
        for (int r2 = 0; r2 < STAGE_ROWS; ++r2) {
          __bf16* dst = wl + (size_t)(gate * 32 + c * STAGE_ROWS + r2) * WROW_STRIDE;
          const float* srow = stage + r2 * HID;
          for (int k = tid; k < HID; k += RTHREADS) dst[k] = (__bf16)srow[k];
        }
        __syncthreads();
      }
    }
  }
#else
  for (int r = 0; r < RTHREADS; ++r) {
    const int grow = (r / 32) * HID + j * 32 + (r & 31);
    const float* src = w_hh + (size_t)grow * HID;
    __bf16* dst = wl + (size_t)r * WROW_STRIDE;
    for (int k = tid; k < HID; k += RTHREADS) dst[k] = (__bf16)src[k];
  }
  __syncthreads();
#endif

  // ---- Initial cell: h=0 => gh = b_hh; gi from Gi row SEQ+dir ----
  ghs[tid] = bhh;
  __syncthreads();
  if (tid < 32) {
    const int col = j * 32 + tid;
    const float* gir = Gi + (size_t)(SEQ + dir) * G3;
    float r = 1.0f / (1.0f + __expf(-(gir[col]       + ghs[tid])));
    float z = 1.0f / (1.0f + __expf(-(gir[HID + col] + ghs[32 + tid])));
    float n = tanhf(gir[2 * HID + col] + r * ghs[64 + tid]);
    hb0[col] = (1.0f - z) * n;
  }
  __threadfence();
  __syncthreads();
  if (tid == 0) {
    atomicAdd(myCnt, 1);
    volatile int* vc = (volatile int*)myCnt;
    while (*vc < RWGS) __builtin_amdgcn_s_sleep(1);
  }
  __syncthreads();
  __threadfence();

  // ---- Sequential scan ----
  for (int t = 0; t < SEQ; ++t) {
    const int trow = (dir == 0) ? t : (SEQ - 1 - t);
    float* hcur = (t & 1) ? hb1 : hb0;
    float* hnxt = (t & 1) ? hb0 : hb1;

    for (int k = tid; k < HID; k += RTHREADS) hs[k] = hcur[k];
    __syncthreads();

    if (t + 1 < SEQ) {  // pull next step's Gi row toward L2/L0
      const int nrow = (dir == 0) ? (t + 1) : (SEQ - 2 - t);
      __builtin_prefetch(&Gi[(size_t)nrow * G3 + wrow], 0, 1);
    }

    // gh[row] = dot(w_hh[row,:], h) + b_hh[row]
    // weights: ds_load_b64 (conflict-free), h: ds_load_b128 broadcast
    const uint2*  wp = (const uint2*)(wl + (size_t)tid * WROW_STRIDE);
    const float4* hp = (const float4*)hs;
    float acc = 0.0f;
    #pragma unroll 4
    for (int q = 0; q < HID / 4; ++q) {
      uint2  w  = wp[q];
      float4 h4 = hp[q];
      acc = fmaf(__uint_as_float((w.x & 0xffffu) << 16), h4.x, acc);
      acc = fmaf(__uint_as_float(w.x & 0xffff0000u),     h4.y, acc);
      acc = fmaf(__uint_as_float((w.y & 0xffffu) << 16), h4.z, acc);
      acc = fmaf(__uint_as_float(w.y & 0xffff0000u),     h4.w, acc);
    }
    ghs[tid] = acc + bhh;
    __syncthreads();

    if (tid < 32) {
      const int col = j * 32 + tid;
      const float* gir = Gi + (size_t)trow * G3;
      float hold = hs[col];
      float r = 1.0f / (1.0f + __expf(-(gir[col]       + ghs[tid])));
      float z = 1.0f / (1.0f + __expf(-(gir[HID + col] + ghs[32 + tid])));
      float n = tanhf(gir[2 * HID + col] + r * ghs[64 + tid]);
      float hnew = (1.0f - z) * n + z * hold;
      hnxt[col] = hnew;
      dout[HID + (size_t)trow * (2 * HID) + dir * HID + col] = hnew;
      if (dir == 1 && trow == 0) dout[col] = hnew;   // h_b = hs_b[0]
    }

    __threadfence();
    __syncthreads();
    if (tid == 0) {
      atomicAdd(myCnt, 1);
      const int expected = RWGS * (t + 2);
      volatile int* vc = (volatile int*)myCnt;
      while (*vc < expected) __builtin_amdgcn_s_sleep(1);
    }
    __syncthreads();
    __threadfence();
  }
}

// ---------------------------------------------------------------------------
extern "C" void kernel_launch(void* const* d_in, const int* in_sizes, int n_in,
                              void* d_out, int out_size, void* d_ws, size_t ws_size,
                              hipStream_t stream) {
  const int*   tokens = (const int*)  d_in[0];
  const float* emb    = (const float*)d_in[1];
  const float* w_ih   = (const float*)d_in[2];
  const float* w_hh   = (const float*)d_in[3];
  const float* b_ih   = (const float*)d_in[4];
  const float* b_hh   = (const float*)d_in[5];
  float* out = (float*)d_out;

  char*  ws   = (char*)d_ws;
  int*   cnt  = (int*)ws;                         // 64 ints
  float* hbuf = (float*)(ws + 256);               // 2 dirs x 2 bufs x 1024 f32
  float* Gi   = (float*)(ws + 256 + 16384);       // 4098 x 3072 f32 (~48 MB)

  init_kernel<<<1, 64, 0, stream>>>(cnt);

  dim3 g1(G3 / BN, (AROWS + BM - 1) / BM);        // 24 x 65
  gi_gemm_kernel<<<g1, 256, 0, stream>>>(tokens, emb, w_ih, b_ih, Gi);

  gru_scan_kernel<<<2 * RWGS, RTHREADS, SCAN_LDS, stream>>>(
      w_hh, b_hh, Gi, hbuf, cnt, out);
}
